// EAC_Module_8349416424139
// MI455X (gfx1250) — matrix operand order repaired and checked by reading, not compiled, not run on hardware
//
#include <hip/hip_runtime.h>
#include <hip/hip_bf16.h>
#include <stdint.h>

typedef __attribute__((ext_vector_type(16))) __bf16 v16bf;
typedef __attribute__((ext_vector_type(8)))  __bf16 v8bf;
typedef __attribute__((ext_vector_type(8)))  float  v8f;

#define IN_DIM 512
#define C_DIM  128
#define NTOK   4096
#define BATCH  8

static __device__ __forceinline__ v8f wmma_bf16(v16bf a, v16bf b, v8f c) {
  // D = A(16x32) x B(32x16) + C, f32 accumulate
  return __builtin_amdgcn_wmma_f32_16x16x32_bf16(false, a, false, b, (short)0, c, false, false);
}

static __device__ __forceinline__ v8f zero8() {
  v8f z;
  #pragma unroll
  for (int i = 0; i < 8; ++i) z[i] = 0.0f;
  return z;
}

// Generic pointer to LDS byte offset (low 32 bits of the flat address).
static __device__ __forceinline__ uint32_t lds_addr(const void* p) {
  return (uint32_t)(uintptr_t)p;
}

// CDNA5 async global->LDS copy, 16B per lane, tracked by ASYNCcnt.
static __device__ __forceinline__ void async_cp16(const __bf16* lds_dst, const __bf16* gsrc) {
  asm volatile("global_load_async_to_lds_b128 %0, %1, off"
               :: "v"(lds_addr(lds_dst)), "v"(gsrc)
               : "memory");
}
static __device__ __forceinline__ void wait_async0() {
  asm volatile("s_wait_asynccnt 0x0" ::: "memory");
}

// A-fragment (16x32, bf16 source, row-major rows=M, stride in elements).
// Lane = h*16+m holds K = {h*8+0..7, 16+h*8+0..7} of row m (per ISA layout).
static __device__ __forceinline__ v16bf load_a16(const __bf16* base, int stride) {
  const int lane = threadIdx.x & 31;
  const int m = lane & 15, h = lane >> 4;
  const __bf16* p0 = base + (size_t)m * stride + h * 8;
  v8bf lo = *(const v8bf*)p0;
  v8bf hi = *(const v8bf*)(p0 + 16);
  v16bf r;
  #pragma unroll
  for (int i = 0; i < 8; ++i) { r[i] = lo[i]; r[i + 8] = hi[i]; }
  return r;
}

// A-fragment from f32 source (weights), converting to bf16.
static __device__ __forceinline__ v16bf load_a16_f32(const float* base, int stride) {
  const int lane = threadIdx.x & 31;
  const int m = lane & 15, h = lane >> 4;
  const float* p0 = base + (size_t)m * stride + h * 8;
  v16bf r;
  #pragma unroll
  for (int i = 0; i < 8; ++i) { r[i] = (__bf16)p0[i]; r[i + 8] = (__bf16)(p0 + 16)[i]; }
  return r;
}

// B-fragment (32x16): memory holds B transposed row-major, i.e. mem[n][k] = B(k,n).
// Lane n%16 takes column n, half h holds K = {h*8+0..7, 16+h*8+0..7} (matches A-fragment).
static __device__ __forceinline__ v16bf load_bT(const __bf16* base, int stride) {
  const int lane = threadIdx.x & 31;
  const int n = lane & 15, h = lane >> 4;
  const __bf16* p0 = base + (size_t)n * stride + h * 8;
  v8bf lo = *(const v8bf*)p0;
  v8bf hi = *(const v8bf*)(p0 + 16);
  v16bf r;
  #pragma unroll
  for (int i = 0; i < 8; ++i) { r[i] = lo[i]; r[i + 8] = hi[i]; }
  return r;
}

// ---------------- Kernel 1: xp = W_in@x + b_in ; q/k/v projections ----------------
// grid = B * (N/128) blocks, 256 threads (8 waves). Wave w owns out-channels [16w,16w+16).
__global__ __launch_bounds__(256) void k1_proj_qkv(
    const float* __restrict__ x,  const float* __restrict__ Win, const float* __restrict__ bin,
    const float* __restrict__ Wq, const float* __restrict__ bq,
    const float* __restrict__ Wk, const float* __restrict__ bk,
    const float* __restrict__ Wv, const float* __restrict__ bv,
    float* __restrict__ xp_ws, __bf16* __restrict__ q_ws,
    __bf16* __restrict__ k_ws, __bf16* __restrict__ vT_ws)
{
  constexpr int XLS = 64;    // token-major X chunk rows (32 valid + pad), 128B rows
  constexpr int XPS = 144;   // token-major Xp rows (128 valid + pad), 288B rows
  __shared__ __bf16 XL[128 * XLS];
  __shared__ __bf16 XP[128 * XPS];

  const int tid  = threadIdx.x;
  const int w    = tid >> 5;
  const int lane = tid & 31;
  const int lm = lane & 15, lh = lane >> 4;
  const int b  = blockIdx.x >> 5;
  const int n0 = (blockIdx.x & 31) * 128;
  const int crow = w * 16;

  // ---- GEMM1: Xp(128x128 tile) = W_in(128x512) @ X(512x128) ----
  v8f acc[8];
  #pragma unroll
  for (int t = 0; t < 8; ++t) acc[t] = zero8();

  #pragma unroll 1
  for (int kb = 0; kb < IN_DIM; kb += 32) {
    __syncthreads();
    {   // cooperative transpose+cvt: XL[n][i] = (bf16) x[b, kb+i, n0+n]
      const int il = tid >> 3;          // 0..31 input-channel
      const int nf = (tid & 7) * 16;    // token offset
      const float* src = x + ((size_t)(b * IN_DIM + kb + il)) * NTOK + n0 + nf;
      #pragma unroll
      for (int j = 0; j < 16; j += 4) {
        float4 v4 = *(const float4*)(src + j);
        XL[(nf + j + 0) * XLS + il] = (__bf16)v4.x;
        XL[(nf + j + 1) * XLS + il] = (__bf16)v4.y;
        XL[(nf + j + 2) * XLS + il] = (__bf16)v4.z;
        XL[(nf + j + 3) * XLS + il] = (__bf16)v4.w;
      }
    }
    __syncthreads();
    v16bf a = load_a16_f32(Win + (size_t)crow * IN_DIM + kb, IN_DIM);
    #pragma unroll
    for (int t = 0; t < 8; ++t) {
      v16bf bf = load_bT(XL + t * 16 * XLS, XLS);
      acc[t] = wmma_bf16(a, bf, acc[t]);
    }
  }

  // bias, write xproj (f32, (C,N) layout) and stage Xp in LDS (bf16 token-major)
  float bias[8];
  #pragma unroll
  for (int r = 0; r < 8; ++r) bias[r] = bin[crow + r + 8 * lh];
  #pragma unroll
  for (int t = 0; t < 8; ++t) {
    #pragma unroll
    for (int r = 0; r < 8; ++r) {
      const float vv = acc[t][r] + bias[r];
      const int c = crow + r + 8 * lh;
      const int n = n0 + t * 16 + lm;
      xp_ws[((size_t)(b * C_DIM + c)) * NTOK + n] = vv;
      XP[(t * 16 + lm) * XPS + c] = (__bf16)vv;
    }
  }
  __syncthreads();

  // ---- q/k/v = W_p(128x128) @ Xp ----
  const float* Wp[3] = { Wq, Wk, Wv };
  const float* bp[3] = { bq, bk, bv };
  #pragma unroll 1
  for (int p = 0; p < 3; ++p) {
    v8f pa[8];
    #pragma unroll
    for (int t = 0; t < 8; ++t) pa[t] = zero8();
    #pragma unroll
    for (int kb = 0; kb < C_DIM; kb += 32) {
      v16bf a = load_a16_f32(Wp[p] + (size_t)crow * C_DIM + kb, C_DIM);
      #pragma unroll
      for (int t = 0; t < 8; ++t) {
        v16bf bf = load_bT(XP + t * 16 * XPS + kb, XPS);
        pa[t] = wmma_bf16(a, bf, pa[t]);
      }
    }
    float pbias[8];
    #pragma unroll
    for (int r = 0; r < 8; ++r) pbias[r] = bp[p][crow + r + 8 * lh];
    #pragma unroll
    for (int t = 0; t < 8; ++t) {
      #pragma unroll
      for (int r = 0; r < 8; ++r) {
        const float vv = pa[t][r] + pbias[r];
        const int c = crow + r + 8 * lh;
        const int n = n0 + t * 16 + lm;
        if (p == 0)      q_ws [((size_t)(b * NTOK + n)) * C_DIM + c] = (__bf16)vv;
        else if (p == 1) k_ws [((size_t)(b * NTOK + n)) * C_DIM + c] = (__bf16)vv;
        else             vT_ws[((size_t)(b * C_DIM + c)) * NTOK + n] = (__bf16)vv;
      }
    }
  }
}

// ---------------- Kernel 2: flash attention + residual ----------------
// grid = B * (N/128), 256 threads; each wave owns 16 query tokens, full C=128.
// K/V tiles (64 keys) staged once per block in LDS via async copies, double-buffered.
__global__ __launch_bounds__(256) void k2_attention(
    const __bf16* __restrict__ q_ws, const __bf16* __restrict__ k_ws,
    const __bf16* __restrict__ vT_ws, const float* __restrict__ xp_ws,
    const float* __restrict__ gamma, float* __restrict__ out)
{
  constexpr int KLS = 136;  // K tile rows (128 valid + pad), 272B rows
  constexpr int VLS = 72;   // V tile rows (64 valid + pad), 144B rows
  constexpr int PBS = 80;   // P row stride (64 valid + pad), 160B rows
  __shared__ __bf16 KL[2][64 * KLS];    // token-major keys:  KL[key][c]
  __shared__ __bf16 VL[2][128 * VLS];   // channel-major vals: VL[c][key]
  __shared__ __bf16 PB[8][16 * PBS];

  const int tid  = threadIdx.x;
  const int w    = tid >> 5;
  const int lane = tid & 31;
  const int lm = lane & 15, lh = lane >> 4;
  const int b  = blockIdx.x >> 5;
  const int q0 = (blockIdx.x & 31) * 128 + w * 16;
  const float g = gamma[0];

  // issue async stage of tile m0 into buffer buf (16KB K + 16KB V per tile)
  auto stage = [&](int buf, int m0) {
    #pragma unroll
    for (int i = 0; i < 4; ++i) {                 // 64 rows x 16 chunks of 16B
      const int idx = tid + i * 256;
      const int row = idx >> 4, col = (idx & 15) * 8;
      async_cp16(&KL[buf][row * KLS + col],
                 k_ws + ((size_t)(b * NTOK + m0 + row)) * C_DIM + col);
    }
    #pragma unroll
    for (int i = 0; i < 4; ++i) {                 // 128 rows x 8 chunks of 16B
      const int idx = tid + i * 256;
      const int row = idx >> 3, col = (idx & 7) * 8;
      async_cp16(&VL[buf][row * VLS + col],
                 vT_ws + ((size_t)(b * C_DIM + row)) * NTOK + m0 + col);
    }
  };

  stage(0, 0);

  // Q fragments for this wave's 16 queries (resident for whole kernel)
  v16bf qa[4];
  #pragma unroll
  for (int cc = 0; cc < 4; ++cc)
    qa[cc] = load_a16(q_ws + ((size_t)(b * NTOK + q0)) * C_DIM + cc * 32, C_DIM);

  v8f O[8];
  #pragma unroll
  for (int ct = 0; ct < 8; ++ct) O[ct] = zero8();
  float mi[8], li[8];
  #pragma unroll
  for (int r = 0; r < 8; ++r) { mi[r] = -1e30f; li[r] = 0.0f; }

  wait_async0();
  __syncthreads();

  #pragma unroll 1
  for (int m0 = 0; m0 < NTOK; m0 += 64) {
    const int buf = (m0 >> 6) & 1;
    if (m0 + 64 < NTOK) stage(buf ^ 1, m0 + 64);   // overlap copy of next tile

    // S = Q K^T  (16 queries x 64 keys), K from LDS
    v8f S[4];
    #pragma unroll
    for (int s = 0; s < 4; ++s) S[s] = zero8();
    #pragma unroll
    for (int s = 0; s < 4; ++s) {
      #pragma unroll
      for (int cc = 0; cc < 4; ++cc) {
        v16bf kf = load_bT(&KL[buf][s * 16 * KLS + cc * 32], KLS);
        S[s] = wmma_bf16(qa[cc], kf, S[s]);
      }
    }
    // online softmax: row stats for row M = r + 8*lh (replicated across 16-lane half)
    float mn[8];
    #pragma unroll
    for (int r = 0; r < 8; ++r) {
      float v = fmaxf(fmaxf(S[0][r], S[1][r]), fmaxf(S[2][r], S[3][r]));
      v = fmaxf(v, __shfl_xor(v, 1));
      v = fmaxf(v, __shfl_xor(v, 2));
      v = fmaxf(v, __shfl_xor(v, 4));
      v = fmaxf(v, __shfl_xor(v, 8));
      mn[r] = fmaxf(mi[r], v);
    }
    #pragma unroll
    for (int r = 0; r < 8; ++r) {
      const float sc = __expf(mi[r] - mn[r]);
      float rs = 0.0f;
      #pragma unroll
      for (int s = 0; s < 4; ++s) {
        const float p = __expf(S[s][r] - mn[r]);
        rs += p;
        PB[w][(r + 8 * lh) * PBS + s * 16 + lm] = (__bf16)p;   // row-major P for A-frag reload
      }
      rs += __shfl_xor(rs, 1);
      rs += __shfl_xor(rs, 2);
      rs += __shfl_xor(rs, 4);
      rs += __shfl_xor(rs, 8);
      li[r] = li[r] * sc + rs;
      mi[r] = mn[r];
      #pragma unroll
      for (int ct = 0; ct < 8; ++ct) O[ct][r] *= sc;
    }
    // O += P V   (P via wave-private LDS; V tile from LDS)
    v16bf pa0 = load_a16(&PB[w][0],  PBS);
    v16bf pa1 = load_a16(&PB[w][32], PBS);
    #pragma unroll
    for (int ct = 0; ct < 8; ++ct) {
      O[ct] = wmma_bf16(pa0, load_bT(&VL[buf][ct * 16 * VLS],      VLS), O[ct]);
      O[ct] = wmma_bf16(pa1, load_bT(&VL[buf][ct * 16 * VLS + 32], VLS), O[ct]);
    }

    wait_async0();     // next tile's async copies complete (per-wave)
    __syncthreads();   // all waves done reading buf + next tile visible block-wide
  }

  // epilogue: out = gamma * O / l + xproj   ((C,N) layout, 2x16B per lane per tile)
  #pragma unroll
  for (int ct = 0; ct < 8; ++ct) {
    const size_t base = ((size_t)(b * C_DIM + ct * 16 + lm)) * NTOK + q0 + 8 * lh;
    float4 x0 = *(const float4*)(xp_ws + base);
    float4 x1 = *(const float4*)(xp_ws + base + 4);
    float4 o0, o1;
    o0.x = g * O[ct][0] / li[0] + x0.x;
    o0.y = g * O[ct][1] / li[1] + x0.y;
    o0.z = g * O[ct][2] / li[2] + x0.z;
    o0.w = g * O[ct][3] / li[3] + x0.w;
    o1.x = g * O[ct][4] / li[4] + x1.x;
    o1.y = g * O[ct][5] / li[5] + x1.y;
    o1.z = g * O[ct][6] / li[6] + x1.z;
    o1.w = g * O[ct][7] / li[7] + x1.w;
    *(float4*)(out + base)     = o0;
    *(float4*)(out + base + 4) = o1;
  }
}

extern "C" void kernel_launch(void* const* d_in, const int* in_sizes, int n_in,
                              void* d_out, int out_size, void* d_ws, size_t ws_size,
                              hipStream_t stream) {
  const float* x    = (const float*)d_in[0];
  const float* Win  = (const float*)d_in[1];
  const float* bin  = (const float*)d_in[2];
  const float* Wq   = (const float*)d_in[3];
  const float* bq   = (const float*)d_in[4];
  const float* Wk   = (const float*)d_in[5];
  const float* bk   = (const float*)d_in[6];
  const float* Wv   = (const float*)d_in[7];
  const float* bv   = (const float*)d_in[8];
  const float* gam  = (const float*)d_in[9];
  float* outp = (float*)d_out;

  // workspace: xproj f32 (16MB) | q bf16 (8MB) | k bf16 (8MB) | vT bf16 (8MB)
  const size_t elems = (size_t)BATCH * C_DIM * NTOK;   // 4,194,304
  float*  xp = (float*)d_ws;
  __bf16* q  = (__bf16*)(xp + elems);
  __bf16* k  = q + elems;
  __bf16* vT = k + elems;

  dim3 grid(BATCH * (NTOK / 128));   // 256 blocks
  dim3 block(256);
  k1_proj_qkv<<<grid, block, 0, stream>>>(x, Win, bin, Wq, bq, Wk, bk, Wv, bv,
                                          xp, q, k, vT);
  k2_attention<<<grid, block, 0, stream>>>(q, k, vT, xp, gam, outp);
}
